// VQVAE_80582176407790
// MI455X (gfx1250) — matrix operand order repaired and checked by reading, not compiled, not run on hardware
//
#include <hip/hip_runtime.h>
#include <hip/hip_bf16.h>

// VQ-VAE nearest-codeword quantization for MI455X (gfx1250, wave32).
// Distances via  ||c||^2 - 2 x.c  computed with split-bf16 WMMA
// (v_wmma_f32_16x16x32_bf16, 3 compensated products per K-step).
// M=32 rows per wave: each 4KB codebook tile load feeds 12 WMMAs.

typedef __attribute__((ext_vector_type(16))) __bf16          v16bf;
typedef __attribute__((ext_vector_type(16))) unsigned short  v16us;
typedef __attribute__((ext_vector_type(8)))  float           v8f;

#define VQ_N 262144
#define VQ_K 1024
#define VQ_D 64
#define VQ_MROWS 32                      // rows of x per wave (2 WMMA row-tiles)
#define VQ_BLOCKS (VQ_N / VQ_MROWS)      // 8192

static __device__ __forceinline__ unsigned short f2bf_rne(float f) {
    unsigned u = __float_as_uint(f);
    unsigned r = u + 0x7fffu + ((u >> 16) & 1u);
    return (unsigned short)(r >> 16);
}
static __device__ __forceinline__ float bf2f(unsigned short h) {
    return __uint_as_float(((unsigned)h) << 16);
}
static __device__ __forceinline__ v16bf load_b16(const unsigned short* p) {
    v16us u = *(const v16us*)p;               // two global_load_b128
    return __builtin_bit_cast(v16bf, u);
}

// ---------------------------------------------------------------------------
// Prep: codebook fp32 -> bf16 hi/lo (row-major), plus exact fp32 ||c||^2.
// One wave per codeword row.
// ---------------------------------------------------------------------------
__global__ __launch_bounds__(32)
void vq_prep_kernel(const float* __restrict__ cb,
                    unsigned short* __restrict__ cb_hi,
                    unsigned short* __restrict__ cb_lo,
                    float* __restrict__ c_sq) {
    const int row  = blockIdx.x;
    const int lane = threadIdx.x;
    float s = 0.f;
#pragma unroll
    for (int i = 0; i < 2; ++i) {
        const int d = lane + 32 * i;
        const float v = cb[(size_t)row * VQ_D + d];
        const unsigned short h = f2bf_rne(v);
        const unsigned short l = f2bf_rne(v - bf2f(h));
        cb_hi[(size_t)row * VQ_D + d] = h;
        cb_lo[(size_t)row * VQ_D + d] = l;
        s += v * v;
    }
#pragma unroll
    for (int m = 16; m > 0; m >>= 1) s += __shfl_xor(s, m, 32);
    if (lane == 0) c_sq[row] = s;
}

// ---------------------------------------------------------------------------
// Main: one wave handles 32 rows of x (two 16-row WMMA tiles) against all
// 1024 codewords. A (2 x 16x64 split-bf16) lives in registers; per 16-column
// codeword tile: 4 contiguous 32B B loads/lane + 12 WMMAs; running argmin.
// ---------------------------------------------------------------------------
__global__ __launch_bounds__(32)
void vq_main_kernel(const float* __restrict__ x,
                    const float* __restrict__ cb,
                    const unsigned short* __restrict__ cb_hi,
                    const unsigned short* __restrict__ cb_lo,
                    const float* __restrict__ c_sq,
                    int* __restrict__ z_out,
                    float* __restrict__ q_out,
                    double* __restrict__ partial) {
    const int lane = threadIdx.x;
    const int hh   = lane >> 4;      // half-wave select
    const int r    = lane & 15;      // row within tile (A) / column slot (B)
    const int base = blockIdx.x * VQ_MROWS;

    // ---- Build two A matrices (each: two K=32 steps), split bf16 hi+lo ---
    // A layout used here: lanes 0-15 row M=r hold K {0..15}; lanes 16-31
    // hold K {16..31}; v16bf elems 0..15 contiguous in K (matches B).
    v16bf a_hi[2][2], a_lo[2][2];    // [row-block][k-step]
#pragma unroll
    for (int rb = 0; rb < 2; ++rb) {
#pragma unroll
        for (int s = 0; s < 2; ++s) {
            const float* xr = x + (size_t)(base + rb * 16 + r) * VQ_D + s * 32 + hh * 16;
            float fv[16];
            *(float4*)&fv[0]  = *(const float4*)(xr + 0);
            *(float4*)&fv[4]  = *(const float4*)(xr + 4);
            *(float4*)&fv[8]  = *(const float4*)(xr + 8);
            *(float4*)&fv[12] = *(const float4*)(xr + 12);
            v16us hv, lv;
#pragma unroll
            for (int j = 0; j < 16; ++j) {
                const unsigned short h = f2bf_rne(fv[j]);
                hv[j] = h;
                lv[j] = f2bf_rne(fv[j] - bf2f(h));
            }
            a_hi[rb][s] = __builtin_bit_cast(v16bf, hv);
            a_lo[rb][s] = __builtin_bit_cast(v16bf, lv);
        }
    }

    float    bestv[2][8];
    unsigned besti[2][8];
#pragma unroll
    for (int rb = 0; rb < 2; ++rb)
#pragma unroll
        for (int j = 0; j < 8; ++j) { bestv[rb][j] = 3.4e38f; besti[rb][j] = 0u; }

    // ---- Sweep the 64 codeword tiles --------------------------------------
    for (int t = 0; t < 64; ++t) {
        const unsigned col = (unsigned)(t * 16 + r);   // this lane's column
        // B layout: lane = column `col`; K contiguous: half-wave offset 16,
        // K-step offset 32 bf16 elements.
        const unsigned short* bh = cb_hi + (size_t)col * VQ_D + hh * 16;
        const unsigned short* bl = cb_lo + (size_t)col * VQ_D + hh * 16;
        const v16bf b_hi0 = load_b16(bh);
        const v16bf b_hi1 = load_b16(bh + 32);
        const v16bf b_lo0 = load_b16(bl);
        const v16bf b_lo1 = load_b16(bl + 32);
        const float csq = c_sq[col];

        v8f acc0 = {}, acc1 = {};
        acc0 = __builtin_amdgcn_wmma_f32_16x16x32_bf16(false, a_hi[0][0], false, b_hi0, (short)0, acc0, false, false);
        acc1 = __builtin_amdgcn_wmma_f32_16x16x32_bf16(false, a_hi[1][0], false, b_hi0, (short)0, acc1, false, false);
        acc0 = __builtin_amdgcn_wmma_f32_16x16x32_bf16(false, a_hi[0][1], false, b_hi1, (short)0, acc0, false, false);
        acc1 = __builtin_amdgcn_wmma_f32_16x16x32_bf16(false, a_hi[1][1], false, b_hi1, (short)0, acc1, false, false);
        acc0 = __builtin_amdgcn_wmma_f32_16x16x32_bf16(false, a_lo[0][0], false, b_hi0, (short)0, acc0, false, false);
        acc1 = __builtin_amdgcn_wmma_f32_16x16x32_bf16(false, a_lo[1][0], false, b_hi0, (short)0, acc1, false, false);
        acc0 = __builtin_amdgcn_wmma_f32_16x16x32_bf16(false, a_lo[0][1], false, b_hi1, (short)0, acc0, false, false);
        acc1 = __builtin_amdgcn_wmma_f32_16x16x32_bf16(false, a_lo[1][1], false, b_hi1, (short)0, acc1, false, false);
        acc0 = __builtin_amdgcn_wmma_f32_16x16x32_bf16(false, a_hi[0][0], false, b_lo0, (short)0, acc0, false, false);
        acc1 = __builtin_amdgcn_wmma_f32_16x16x32_bf16(false, a_hi[1][0], false, b_lo0, (short)0, acc1, false, false);
        acc0 = __builtin_amdgcn_wmma_f32_16x16x32_bf16(false, a_hi[0][1], false, b_lo1, (short)0, acc0, false, false);
        acc1 = __builtin_amdgcn_wmma_f32_16x16x32_bf16(false, a_hi[1][1], false, b_lo1, (short)0, acc1, false, false);

#pragma unroll
        for (int j = 0; j < 8; ++j) {
            const float d0 = __builtin_fmaf(-2.f, acc0[j], csq);
            const bool b0 = d0 < bestv[0][j];          // strict < keeps lowest idx
            besti[0][j] = b0 ? col : besti[0][j];
            bestv[0][j] = b0 ? d0  : bestv[0][j];
            const float d1 = __builtin_fmaf(-2.f, acc1[j], csq);
            const bool b1 = d1 < bestv[1][j];
            besti[1][j] = b1 ? col : besti[1][j];
            bestv[1][j] = b1 ? d1  : bestv[1][j];
        }
    }

    // ---- Cross-lane argmin within each 16-lane half (rows j / j+8) --------
#pragma unroll
    for (int rb = 0; rb < 2; ++rb) {
#pragma unroll
        for (int j = 0; j < 8; ++j) {
#pragma unroll
            for (int m = 1; m <= 8; m <<= 1) {
                const float    ov = __shfl_xor(bestv[rb][j], m, 32);
                const unsigned oi = (unsigned)__shfl_xor((int)besti[rb][j], m, 32);
                const bool take = (ov < bestv[rb][j]) ||
                                  (ov == bestv[rb][j] && oi < besti[rb][j]);
                bestv[rb][j] = take ? ov : bestv[rb][j];
                besti[rb][j] = take ? oi : besti[rb][j];
            }
        }
    }

    __shared__ int zidx[VQ_MROWS];
    if (lane == 0) {
#pragma unroll
        for (int rb = 0; rb < 2; ++rb)
#pragma unroll
            for (int j = 0; j < 8; ++j) {
                zidx[rb * 16 + j] = (int)besti[rb][j];
                z_out[base + rb * 16 + j] = (int)besti[rb][j];
            }
    }
    if (lane == 16) {
#pragma unroll
        for (int rb = 0; rb < 2; ++rb)
#pragma unroll
            for (int j = 0; j < 8; ++j) {
                zidx[rb * 16 + 8 + j] = (int)besti[rb][j];
                z_out[base + rb * 16 + 8 + j] = (int)besti[rb][j];
            }
    }
    __syncthreads();

    // ---- Gather q (fp32 codebook), write out, accumulate (q-x)^2 ---------
    double lsum = 0.0;
#pragma unroll
    for (int it = 0; it < (VQ_MROWS * VQ_D) / 32; ++it) {
        const int e   = it * 32 + lane;
        const int row = e >> 6;
        const int d   = e & 63;
        const float qv = cb[(size_t)zidx[row] * VQ_D + d];
        const float xv = x[(size_t)(base + row) * VQ_D + d];
        q_out[(size_t)(base + row) * VQ_D + d] = qv;
        const float df = qv - xv;
        lsum += (double)df * (double)df;
    }
#pragma unroll
    for (int m = 16; m > 0; m >>= 1) lsum += __shfl_xor(lsum, m, 32);
    if (lane == 0) partial[blockIdx.x] = lsum;
}

// ---------------------------------------------------------------------------
// Deterministic fixed-order loss reduction: loss = sum / (N*D), both outputs.
// ---------------------------------------------------------------------------
__global__ __launch_bounds__(256)
void vq_loss_kernel(const double* __restrict__ partial, float* __restrict__ loss_out) {
    __shared__ double sm[256];
    const int tid = threadIdx.x;
    double s = 0.0;
    for (int i = tid; i < VQ_BLOCKS; i += 256) s += partial[i];
    sm[tid] = s;
    __syncthreads();
    for (int o = 128; o > 0; o >>= 1) {
        if (tid < o) sm[tid] += sm[tid + o];
        __syncthreads();
    }
    if (tid == 0) {
        const float loss = (float)(sm[0] / ((double)VQ_N * (double)VQ_D));
        loss_out[0] = loss;   // vq_loss
        loss_out[1] = loss;   // commitment_loss (identical forward value)
    }
}

extern "C" void kernel_launch(void* const* d_in, const int* in_sizes, int n_in,
                              void* d_out, int out_size, void* d_ws, size_t ws_size,
                              hipStream_t stream) {
    (void)in_sizes; (void)n_in; (void)out_size; (void)ws_size;
    const float* x  = (const float*)d_in[0];   // [N, 64]
    const float* cb = (const float*)d_in[1];   // [1024, 64]

    // Workspace layout (~324 KB):
    double*         partial = (double*)d_ws;                               // 8192 * 8B
    float*          c_sq    = (float*)((char*)d_ws + VQ_BLOCKS * 8);       // 4 KB
    unsigned short* cb_hi   = (unsigned short*)((char*)c_sq + VQ_K * 4);   // 128 KB
    unsigned short* cb_lo   = cb_hi + (size_t)VQ_K * VQ_D;                 // 128 KB

    // Output layout: Z [N] (int bits), q [N*64] f32, vq_loss, commitment_loss.
    int*   z_out    = (int*)d_out;
    float* q_out    = (float*)d_out + VQ_N;
    float* loss_out = (float*)d_out + VQ_N + (size_t)VQ_N * VQ_D;

    vq_prep_kernel<<<VQ_K, 32, 0, stream>>>(cb, cb_hi, cb_lo, c_sq);
    vq_main_kernel<<<VQ_BLOCKS, 32, 0, stream>>>(x, cb, cb_hi, cb_lo, c_sq,
                                                 z_out, q_out, partial);
    vq_loss_kernel<<<1, 256, 0, stream>>>(partial, loss_out);
}
